// Decoder_75479755260139
// MI455X (gfx1250) — matrix operand order repaired and checked by reading, not compiled, not run on hardware
//
#include <hip/hip_runtime.h>
#include <math.h>

// ---------------------------------------------------------------------------
// CDNA5 (gfx1250) transformer decoder forward.
// All matmuls via v_wmma_f32_16x16x32_bf16 (wave32 WMMA, fp32 accumulate).
// Fragments are fed by 16-byte ds_load_b128 pairs (layouts arranged so every
// per-lane fragment is contiguous in LDS).
// ---------------------------------------------------------------------------

typedef __bf16 bf16;
typedef __attribute__((ext_vector_type(16))) __bf16 v16bf;
typedef __attribute__((ext_vector_type(8)))  __bf16 v8bf;
typedef __attribute__((ext_vector_type(8)))  float  v8f;

#define L_LAYERS 6
#define D_MODEL  1024
#define N_HEADS  16
#define SEQ      2048
#define BATCH    4
#define DKH      64
#define MTOK     (BATCH * SEQ)       // 8192 token rows

__device__ inline v8f wmma_bf16(v16bf a, v16bf b, v8f c) {
  return __builtin_amdgcn_wmma_f32_16x16x32_bf16(
      /*neg_a=*/false, a, /*neg_b=*/false, b,
      /*c_mod=*/(short)0, c, /*reuse_a=*/false, /*reuse_b=*/false);
}

// --- fragment gathers from LDS per CDNA5 ISA 7.12.2 wave32 layouts ----------
// A (16x32 bf16): lane L holds row M = L&15.
//   VGPR0..3 -> K = 8*(L>>4)+{0..7}; VGPR4..7 -> K = 16+8*(L>>4)+{0..7}
// Both runs are 8 contiguous bf16 -> two ds_load_b128.
// stride must be a multiple of 8 elements; base 16B-aligned.
__device__ inline v16bf frag_load_A(const bf16* p, int stride) {
  int lane = threadIdx.x & 31;
  int row  = lane & 15;
  int hi8  = (lane >> 4) << 3;        // 0 or 8
  const bf16* q = p + row * stride + hi8;
  v8bf lo = *(const v8bf*)(q);
  v8bf hc = *(const v8bf*)(q + 16);
  return __builtin_shufflevector(lo, hc, 0, 1, 2, 3, 4, 5, 6, 7,
                                 8, 9, 10, 11, 12, 13, 14, 15);
}

// B (k x n), staged TRANSPOSED in LDS as [n][k]: lane L holds col N = L&15,
// K = 8*(L>>4)+{0..7} and 16+8*(L>>4)+{0..7} (same as A) -> two ds_load_b128.
__device__ inline v16bf frag_load_BT(const bf16* p, int stride) {
  int lane = threadIdx.x & 31;
  int col  = lane & 15;
  int kb   = (lane >> 4) << 3;        // 0 or 8
  const bf16* q = p + col * stride + kb;
  v8bf lo = *(const v8bf*)(q);
  v8bf hc = *(const v8bf*)(q + 16);
  return __builtin_shufflevector(lo, hc, 0, 1, 2, 3, 4, 5, 6, 7,
                                 8, 9, 10, 11, 12, 13, 14, 15);
}
// C/D (16x16 f32): element (vgpr r, lane L) = (row r + 8*(L>>4), col L&15)

// ---------------------------------------------------------------------------
// GEMM: C[M,N] = A[M,K] @ B[K,N] + bias, with B supplied TRANSPOSED [N][K].
// Block: 256 thr = 8 waves (4 along M x 2 along N); wave tile 32x64.
// ---------------------------------------------------------------------------
#define BM 128
#define BN 128
#define BK 32
#define PAD 8                          // LDS row stride BK+PAD = 40 elements

__global__ __launch_bounds__(256) void gemm_bias_kernel(
    const bf16* __restrict__ A, const bf16* __restrict__ Bt,
    const float* __restrict__ bias, float* __restrict__ C,
    int N, int K) {
  __shared__ alignas(16) bf16 sA[BM][BK + PAD];
  __shared__ alignas(16) bf16 sB[BN][BK + PAD];   // [n][k]

  int tid  = threadIdx.x;
  int lane = tid & 31;
  int wave = tid >> 5;
  int wm   = wave & 3;
  int wn   = wave >> 2;
  int bm   = blockIdx.y * BM;
  int bn   = blockIdx.x * BN;

  v8f acc[2][4] = {};

  for (int k0 = 0; k0 < K; k0 += BK) {
    // A tile: 128 rows x 32 k  = 512 x v8bf, 2 per thread
#pragma unroll
    for (int ii = 0; ii < 2; ++ii) {
      int i = tid + ii * 256;
      int r = i >> 2, c = (i & 3) * 8;
      const bf16* g = &A[(size_t)(bm + r) * K + k0 + c];
      *(v8bf*)&sA[r][c] = *(const v8bf*)g;
      __builtin_prefetch(g + BK, 0, 3);           // next k-slice
    }
    // B tile (transposed): 128 n-rows x 32 k = 512 x v8bf, 2 per thread
#pragma unroll
    for (int ii = 0; ii < 2; ++ii) {
      int i = tid + ii * 256;
      int r = i >> 2, c = (i & 3) * 8;
      const bf16* g = &Bt[(size_t)(bn + r) * K + k0 + c];
      *(v8bf*)&sB[r][c] = *(const v8bf*)g;
      __builtin_prefetch(g + BK, 0, 3);
    }
    __syncthreads();

    v16bf a0 = frag_load_A(&sA[wm * 32][0],      BK + PAD);
    v16bf a1 = frag_load_A(&sA[wm * 32 + 16][0], BK + PAD);
    v16bf b0 = frag_load_BT(&sB[wn * 64 +  0][0], BK + PAD);
    v16bf b1 = frag_load_BT(&sB[wn * 64 + 16][0], BK + PAD);
    v16bf b2 = frag_load_BT(&sB[wn * 64 + 32][0], BK + PAD);
    v16bf b3 = frag_load_BT(&sB[wn * 64 + 48][0], BK + PAD);
    acc[0][0] = wmma_bf16(a0, b0, acc[0][0]);
    acc[0][1] = wmma_bf16(a0, b1, acc[0][1]);
    acc[0][2] = wmma_bf16(a0, b2, acc[0][2]);
    acc[0][3] = wmma_bf16(a0, b3, acc[0][3]);
    acc[1][0] = wmma_bf16(a1, b0, acc[1][0]);
    acc[1][1] = wmma_bf16(a1, b1, acc[1][1]);
    acc[1][2] = wmma_bf16(a1, b2, acc[1][2]);
    acc[1][3] = wmma_bf16(a1, b3, acc[1][3]);
    __syncthreads();
  }

  int coln  = lane & 15;
  int rbase = (lane >> 4) << 3;
#pragma unroll
  for (int t = 0; t < 4; ++t) {
    int gc = bn + wn * 64 + t * 16 + coln;
    float bv = bias[gc];
#pragma unroll
    for (int r = 0; r < 8; ++r) {
      int gr0 = bm + wm * 32 + rbase + r;
      C[(size_t)gr0 * N + gc]        = acc[0][t][r] + bv;
      C[(size_t)(gr0 + 16) * N + gc] = acc[1][t][r] + bv;
    }
  }
}

// ---------------------------------------------------------------------------
// Flash attention. Q/K layout [B*H, S, DK]; V layout TRANSPOSED [B*H, DK, S]
// so the P.V B-operand ([d][key]) stages contiguous. Q pre-scaled 1/sqrt(DK).
// Output O: [B, S, D] bf16 (heads concatenated).
// ---------------------------------------------------------------------------
__global__ __launch_bounds__(128) void flash_attn_kernel(
    const bf16* __restrict__ Q, const bf16* __restrict__ K,
    const bf16* __restrict__ Vt, bf16* __restrict__ O) {
  __shared__ alignas(16) bf16 sQ[64][DKH + 8];     // stride 72
  __shared__ alignas(16) bf16 sK[32][DKH + 8];     // [key][d] == BT for scores
  __shared__ alignas(16) bf16 sVT[DKH][32 + 8];    // [d][key] == BT for P.V
  __shared__ alignas(16) bf16 sP[4][16][32 + 8];   // per-wave P (A-layout src)

  int tid  = threadIdx.x;
  int lane = tid & 31;
  int wave = tid >> 5;
  int qb   = blockIdx.x * 64;
  int bh   = blockIdx.y;
  int b    = bh / N_HEADS, h = bh % N_HEADS;

  // Q tile: 64 rows x 64 d = 512 v8bf, 4 per thread
#pragma unroll
  for (int ii = 0; ii < 4; ++ii) {
    int i = tid + ii * 128;
    int r = i >> 3, c = (i & 7) * 8;
    *(v8bf*)&sQ[r][c] = *(const v8bf*)&Q[((size_t)bh * SEQ + qb + r) * DKH + c];
  }
  __syncthreads();

  v16bf qa0 = frag_load_A(&sQ[wave * 16][0],  DKH + 8);  // d 0..31
  v16bf qa1 = frag_load_A(&sQ[wave * 16][32], DKH + 8);  // d 32..63

  v8f o0 = {}, o1 = {}, o2 = {}, o3 = {};
  float m[8], l[8];
#pragma unroll
  for (int r = 0; r < 8; ++r) { m[r] = -3.0e38f; l[r] = 0.0f; }

  int hi   = lane >> 4;
  int coln = lane & 15;
  int nchunks = (qb + 64) / 32;       // causal: keys [0, qb+64)

  for (int kc = 0; kc < nchunks; ++kc) {
    int kb = kc * 32;
    __syncthreads();                  // previous chunk fully consumed
    // K chunk: 32 keys x 64 d = 256 v8bf, 2 per thread
#pragma unroll
    for (int ii = 0; ii < 2; ++ii) {
      int i = tid + ii * 128;
      int r = i >> 3, c = (i & 7) * 8;
      *(v8bf*)&sK[r][c] =
          *(const v8bf*)&K[((size_t)bh * SEQ + kb + r) * DKH + c];
    }
    // V chunk (transposed source): 64 d x 32 keys = 256 v8bf, 2 per thread
#pragma unroll
    for (int ii = 0; ii < 2; ++ii) {
      int i = tid + ii * 128;
      int r = i >> 2, c = (i & 3) * 8;
      *(v8bf*)&sVT[r][c] =
          *(const v8bf*)&Vt[((size_t)bh * DKH + r) * SEQ + kb + c];
    }
    __syncthreads();

    v8f s0 = {}, s1 = {};
    s0 = wmma_bf16(qa0, frag_load_BT(&sK[0][0],   DKH + 8), s0);
    s0 = wmma_bf16(qa1, frag_load_BT(&sK[0][32],  DKH + 8), s0);
    s1 = wmma_bf16(qa0, frag_load_BT(&sK[16][0],  DKH + 8), s1);
    s1 = wmma_bf16(qa1, frag_load_BT(&sK[16][32], DKH + 8), s1);

    float alpha[8];
#pragma unroll
    for (int r = 0; r < 8; ++r) {
      int qg  = qb + wave * 16 + 8 * hi + r;
      int kg0 = kb + coln, kg1 = kb + 16 + coln;
      bool ok0 = (kg0 <= qg), ok1 = (kg1 <= qg);
      float v0 = ok0 ? s0[r] : -3.0e38f;
      float v1 = ok1 ? s1[r] : -3.0e38f;
      float mr = fmaxf(v0, v1);
      mr = fmaxf(mr, __shfl_xor(mr, 1, 32));
      mr = fmaxf(mr, __shfl_xor(mr, 2, 32));
      mr = fmaxf(mr, __shfl_xor(mr, 4, 32));
      mr = fmaxf(mr, __shfl_xor(mr, 8, 32));   // reduce over the 16-lane half
      float mn = fmaxf(m[r], mr);
      float p0 = ok0 ? __expf(s0[r] - mn) : 0.0f;
      float p1 = ok1 ? __expf(s1[r] - mn) : 0.0f;
      float rs = p0 + p1;
      rs += __shfl_xor(rs, 1, 32);
      rs += __shfl_xor(rs, 2, 32);
      rs += __shfl_xor(rs, 4, 32);
      rs += __shfl_xor(rs, 8, 32);
      alpha[r] = __expf(m[r] - mn);
      l[r] = l[r] * alpha[r] + rs;
      m[r] = mn;
      sP[wave][8 * hi + r][coln]      = (bf16)p0;
      sP[wave][8 * hi + r][16 + coln] = (bf16)p1;
    }
#pragma unroll
    for (int r = 0; r < 8; ++r) {
      o0[r] *= alpha[r]; o1[r] *= alpha[r];
      o2[r] *= alpha[r]; o3[r] *= alpha[r];
    }

    v16bf pa = frag_load_A(&sP[wave][0][0], 32 + 8);   // 16q x 32keys
    o0 = wmma_bf16(pa, frag_load_BT(&sVT[ 0][0], 32 + 8), o0);
    o1 = wmma_bf16(pa, frag_load_BT(&sVT[16][0], 32 + 8), o1);
    o2 = wmma_bf16(pa, frag_load_BT(&sVT[32][0], 32 + 8), o2);
    o3 = wmma_bf16(pa, frag_load_BT(&sVT[48][0], 32 + 8), o3);
  }

#pragma unroll
  for (int r = 0; r < 8; ++r) {
    int q = qb + wave * 16 + 8 * hi + r;
    float inv = 1.0f / l[r];
    size_t base = ((size_t)(b * SEQ + q)) * D_MODEL + h * DKH;
    O[base +  0 + coln] = (bf16)(o0[r] * inv);
    O[base + 16 + coln] = (bf16)(o1[r] * inv);
    O[base + 32 + coln] = (bf16)(o2[r] * inv);
    O[base + 48 + coln] = (bf16)(o3[r] * inv);
  }
}

// ---------------------------------------------------------------------------
// Elementwise / row kernels
// ---------------------------------------------------------------------------
// Weights: fp32 [L][K][N] -> bf16 TRANSPOSED [L][N][K]
__global__ void convert_wT_kernel(const float* __restrict__ src,
                                  bf16* __restrict__ dst, int K, int N,
                                  size_t total) {
  size_t i = (size_t)blockIdx.x * 256 + threadIdx.x;
  if (i >= total) return;
  size_t kn = (size_t)K * N;
  size_t l = i / kn, rem = i - l * kn;
  int k = (int)(rem / N), n = (int)(rem - (size_t)k * N);
  dst[l * kn + (size_t)n * K + k] = (bf16)src[i];
}

__global__ void embed_kernel(const int* __restrict__ tokens,
                             const float* __restrict__ emb,
                             const float* __restrict__ pos,
                             float* __restrict__ xf, bf16* __restrict__ xb) {
  size_t i = (size_t)blockIdx.x * 256 + threadIdx.x;
  if (i >= (size_t)MTOK * D_MODEL) return;
  int d = (int)(i % D_MODEL);
  size_t bs = i / D_MODEL;
  int s = (int)(bs % SEQ);
  int t = tokens[bs];
  float v = emb[(size_t)t * D_MODEL + d] + pos[(size_t)s * D_MODEL + d];
  xf[i] = v;
  xb[i] = (bf16)v;
}

// qkv fp32 [B,S,3D] -> q*0.125, k bf16 [B*H,S,DK];  v bf16 TRANSPOSED [B*H,DK,S]
__global__ void split_qkv_kernel(const float* __restrict__ qkv,
                                 bf16* __restrict__ q, bf16* __restrict__ k,
                                 bf16* __restrict__ vT) {
  size_t i = (size_t)blockIdx.x * 256 + threadIdx.x;
  if (i >= (size_t)MTOK * D_MODEL) return;
  size_t bs = i / D_MODEL;
  int hd = (int)(i % D_MODEL);
  int h = hd >> 6, d = hd & 63;
  int b = (int)(bs / SEQ), s = (int)(bs % SEQ);
  size_t src = bs * (size_t)(3 * D_MODEL);
  size_t bhs = ((size_t)(b * N_HEADS + h)) * SEQ;
  q[(bhs + s) * DKH + d] = (bf16)(qkv[src + hd] * 0.125f);   // 1/sqrt(64)
  k[(bhs + s) * DKH + d] = (bf16)(qkv[src + D_MODEL + hd]);
  vT[((size_t)(b * N_HEADS + h) * DKH + d) * SEQ + s] =
      (bf16)(qkv[src + 2 * D_MODEL + hd]);
}

__device__ inline float block_reduce_sum(float v, float* red) {
  int tid = threadIdx.x;
  red[tid] = v;
  __syncthreads();
  for (int off = 128; off > 0; off >>= 1) {
    if (tid < off) red[tid] += red[tid + off];
    __syncthreads();
  }
  float r = red[0];
  __syncthreads();
  return r;
}

// x = LN(proj + x) ; writes xf (in place) and xb
__global__ __launch_bounds__(256) void residual_ln_kernel(
    const float* __restrict__ proj, float* __restrict__ xf,
    bf16* __restrict__ xb, const float* __restrict__ g,
    const float* __restrict__ bb) {
  __shared__ float red[256];
  size_t row = blockIdx.x;
  int tid = threadIdx.x;
  const float* pr = proj + row * D_MODEL;
  float* xr = xf + row * D_MODEL;
  float vals[4], s = 0.0f;
#pragma unroll
  for (int i = 0; i < 4; ++i) {
    int c = tid + i * 256;
    vals[i] = pr[c] + xr[c];
    s += vals[i];
  }
  float mu = block_reduce_sum(s, red) * (1.0f / D_MODEL);
  float s2 = 0.0f;
#pragma unroll
  for (int i = 0; i < 4; ++i) { float d = vals[i] - mu; s2 += d * d; }
  float var = block_reduce_sum(s2, red) * (1.0f / D_MODEL);
  float rinv = rsqrtf(var + 1e-5f);
#pragma unroll
  for (int i = 0; i < 4; ++i) {
    int c = tid + i * 256;
    float y = (vals[i] - mu) * rinv * g[c] + bb[c];
    xr[c] = y;
    xb[row * D_MODEL + c] = (bf16)y;
  }
}

// x = LN(h * gelu_exact(gate)) from hg [M, 2D]; writes dstf and xb
__global__ __launch_bounds__(256) void geglu_ln_kernel(
    const float* __restrict__ hg, float* __restrict__ dstf,
    bf16* __restrict__ xb, const float* __restrict__ g,
    const float* __restrict__ bb) {
  __shared__ float red[256];
  size_t row = blockIdx.x;
  int tid = threadIdx.x;
  const float* hr = hg + row * (size_t)(2 * D_MODEL);
  float vals[4], s = 0.0f;
#pragma unroll
  for (int i = 0; i < 4; ++i) {
    int c = tid + i * 256;
    float hv = hr[c], gt = hr[D_MODEL + c];
    float ge = 0.5f * gt * (1.0f + erff(gt * 0.7071067811865475f));
    vals[i] = hv * ge;
    s += vals[i];
  }
  float mu = block_reduce_sum(s, red) * (1.0f / D_MODEL);
  float s2 = 0.0f;
#pragma unroll
  for (int i = 0; i < 4; ++i) { float d = vals[i] - mu; s2 += d * d; }
  float var = block_reduce_sum(s2, red) * (1.0f / D_MODEL);
  float rinv = rsqrtf(var + 1e-5f);
#pragma unroll
  for (int i = 0; i < 4; ++i) {
    int c = tid + i * 256;
    float y = (vals[i] - mu) * rinv * g[c] + bb[c];
    dstf[row * D_MODEL + c] = y;
    xb[row * D_MODEL + c] = (bf16)y;
  }
}

// ---------------------------------------------------------------------------
// Host side
// ---------------------------------------------------------------------------
extern "C" void kernel_launch(void* const* d_in, const int* in_sizes, int n_in,
                              void* d_out, int out_size, void* d_ws,
                              size_t ws_size, hipStream_t stream) {
  const int*   tokens = (const int*)d_in[0];
  const float* emb    = (const float*)d_in[1];
  const float* pos    = (const float*)d_in[2];
  const float* qkv_w  = (const float*)d_in[3];
  const float* qkv_b  = (const float*)d_in[4];
  const float* out_w  = (const float*)d_in[5];
  const float* out_b  = (const float*)d_in[6];
  const float* ln1_g  = (const float*)d_in[7];
  const float* ln1_b  = (const float*)d_in[8];
  const float* ff_w   = (const float*)d_in[9];
  const float* ff_b   = (const float*)d_in[10];
  const float* ln2_g  = (const float*)d_in[11];
  const float* ln2_b  = (const float*)d_in[12];
  (void)in_sizes; (void)n_in; (void)out_size; (void)ws_size;

  char* ws = (char*)d_ws;
  size_t off = 0;
  auto carve = [&](size_t bytes) -> char* {
    char* p = ws + off;
    off += (bytes + 255) & ~(size_t)255;
    return p;
  };

  const size_t ND = (size_t)MTOK * D_MODEL;          // 8M token-elements
  float* xf    = (float*)carve(ND * 4);
  bf16*  xb    = (bf16*) carve(ND * 2);
  float* qkvf  = (float*)carve(ND * 3 * 4);
  bf16*  qb16  = (bf16*) carve(ND * 2);
  bf16*  kb16  = (bf16*) carve(ND * 2);
  bf16*  vT16  = (bf16*) carve(ND * 2);
  bf16*  ob16  = (bf16*) carve(ND * 2);
  float* projf = (float*)carve(ND * 4);
  float* hgf   = (float*)carve(ND * 2 * 4);
  bf16*  wqT   = (bf16*) carve((size_t)L_LAYERS * D_MODEL * 3 * D_MODEL * 2);
  bf16*  woT   = (bf16*) carve((size_t)L_LAYERS * D_MODEL * D_MODEL * 2);
  bf16*  wfT   = (bf16*) carve((size_t)L_LAYERS * D_MODEL * 2 * D_MODEL * 2);

  // one-time weight convert + transpose: [L][K][N] f32 -> [L][N][K] bf16
  auto cvtT = [&](const float* s, bf16* d, int K, int N) {
    size_t n = (size_t)L_LAYERS * K * N;
    convert_wT_kernel<<<(unsigned)((n + 255) / 256), 256, 0, stream>>>(s, d, K,
                                                                       N, n);
  };
  cvtT(qkv_w, wqT, D_MODEL, 3 * D_MODEL);
  cvtT(out_w, woT, D_MODEL, D_MODEL);
  cvtT(ff_w,  wfT, D_MODEL, 2 * D_MODEL);

  embed_kernel<<<(unsigned)(ND / 256), 256, 0, stream>>>(tokens, emb, pos, xf,
                                                         xb);

  for (int l = 0; l < L_LAYERS; ++l) {
    // QKV projection: [8192,1024] x [1024,3072]
    {
      dim3 grid(3 * D_MODEL / BN, MTOK / BM);
      gemm_bias_kernel<<<grid, 256, 0, stream>>>(
          xb, wqT + (size_t)l * D_MODEL * 3 * D_MODEL,
          qkv_b + (size_t)l * 3 * D_MODEL, qkvf, 3 * D_MODEL, D_MODEL);
    }
    split_qkv_kernel<<<(unsigned)(ND / 256), 256, 0, stream>>>(qkvf, qb16,
                                                               kb16, vT16);
    // Flash attention: grid = (query tiles, B*H)
    {
      dim3 grid(SEQ / 64, BATCH * N_HEADS);
      flash_attn_kernel<<<grid, 128, 0, stream>>>(qb16, kb16, vT16, ob16);
    }
    // Output projection: [8192,1024] x [1024,1024]
    {
      dim3 grid(D_MODEL / BN, MTOK / BM);
      gemm_bias_kernel<<<grid, 256, 0, stream>>>(
          ob16, woT + (size_t)l * D_MODEL * D_MODEL,
          out_b + (size_t)l * D_MODEL, projf, D_MODEL, D_MODEL);
    }
    residual_ln_kernel<<<MTOK, 256, 0, stream>>>(
        projf, xf, xb, ln1_g + (size_t)l * D_MODEL, ln1_b + (size_t)l * D_MODEL);
    // GEGLU FF: [8192,1024] x [1024,2048]
    {
      dim3 grid(2 * D_MODEL / BN, MTOK / BM);
      gemm_bias_kernel<<<grid, 256, 0, stream>>>(
          xb, wfT + (size_t)l * D_MODEL * 2 * D_MODEL,
          ff_b + (size_t)l * 2 * D_MODEL, hgf, 2 * D_MODEL, D_MODEL);
    }
    float* dstf = (l == L_LAYERS - 1) ? (float*)d_out : xf;
    geglu_ln_kernel<<<MTOK, 256, 0, stream>>>(
        hgf, dstf, xb, ln2_g + (size_t)l * D_MODEL, ln2_b + (size_t)l * D_MODEL);
  }
}
